// FairseqDecoder_73229192397490
// MI455X (gfx1250) — matrix operand order repaired and checked by reading, not compiled, not run on hardware
//
#include <hip/hip_runtime.h>
#include <hip/hip_bf16.h>

#define SEQLEN 64
#define BATCH  32
#define VOCAB  32000
#define EMB    512
#define HID    1024

typedef __attribute__((ext_vector_type(16))) __bf16 v16bf;
typedef __attribute__((ext_vector_type(8)))  __bf16 v8bf;
typedef __attribute__((ext_vector_type(8)))  float  v8f;

// ---------------------------------------------------------------------------
// Fragment loaders for CDNA5 WMMA (CDNA5 ISA); A and B share this code's contiguous k order
//   A 16x32 bf16 : lane l (m = l&15, lh = l>>4) holds K = kb+16lh .. +15
//                  -> two contiguous 16B runs at +0 / +8.
//   B 32x16 bf16 : lane l (n = l&15, lh = l>>4) holds K = kb+16lh .. +15
//                  -> two contiguous 16B runs at +0 / +8.
//   C/D 16x16 f32: element r -> row m = 8*(l>>4) + r, col n = l&15.
// ---------------------------------------------------------------------------
__device__ __forceinline__ v16bf load_frag16(const __bf16* p0, const __bf16* p1) {
  v8bf lo = *reinterpret_cast<const v8bf*>(p0);
  v8bf hi = *reinterpret_cast<const v8bf*>(p1);
  return __builtin_shufflevector(lo, hi, 0,1,2,3,4,5,6,7,8,9,10,11,12,13,14,15);
}

__device__ __forceinline__ v8f wmma_bf16(v8f acc, v16bf a, v16bf b) {
  return __builtin_amdgcn_wmma_f32_16x16x32_bf16(false, a, false, b,
                                                 (short)0, acc, false, false);
}

__device__ __forceinline__ float sigmoidf_(float x) {
  return 1.0f / (1.0f + __expf(-x));
}
__device__ __forceinline__ float tanhf_(float x) {
  // 1 - 2/(e^{2x}+1): saturates correctly at +-1 without inf/inf NaN
  return 1.0f - 2.0f / (__expf(2.0f * x) + 1.0f);
}

// ---------------------------------------------------------------------------
// fp32 -> bf16 weight conversion
// ---------------------------------------------------------------------------
__global__ void f32_to_bf16_kernel(const float* __restrict__ in,
                                   __bf16* __restrict__ out, long n) {
  long t = (long)blockIdx.x * blockDim.x + threadIdx.x;
  if (t < n) out[t] = (__bf16)in[t];
}

// Gather embeddings for all timesteps: xemb[s][b][e] = emb[idx[s][b]][e]
__global__ void embed_kernel(const int* __restrict__ idx,
                             const float* __restrict__ emb,
                             __bf16* __restrict__ xemb) {
  long t = (long)blockIdx.x * blockDim.x + threadIdx.x;
  if (t >= (long)SEQLEN * BATCH * EMB) return;
  int e = (int)(t & (EMB - 1));
  long rem = t >> 9;               // / EMB
  int b = (int)(rem & (BATCH - 1));
  int s = (int)(rem >> 5);         // / BATCH
  int row = idx[s * BATCH + b];
  xemb[t] = (__bf16)emb[(size_t)row * EMB + e];
}

// h0/h1 <- enc_hidden, ctx <- 0
__global__ void init_state_kernel(const float* __restrict__ ench,
                                  float* __restrict__ h0f, __bf16* __restrict__ h0b,
                                  float* __restrict__ h1f, __bf16* __restrict__ h1b,
                                  __bf16* __restrict__ ctx0) {
  int t = blockIdx.x * blockDim.x + threadIdx.x;
  if (t >= BATCH * HID) return;
  float a = ench[t];
  float c = ench[BATCH * HID + t];
  h0f[t] = a; h0b[t] = (__bf16)a;
  h1f[t] = c; h1b[t] = (__bf16)c;
  ctx0[t] = (__bf16)0.0f;
}

// ---------------------------------------------------------------------------
// Fused GRU cell for one timestep.
// Input x = [A0 (lenA0 cols) | A1] (bf16, row-major, B=32 rows).
//   gi = x @ Wih^T , gh = h @ Whh^T  (Wih: [3H, Kin], Whh: [3H, H], bf16)
// Gate math in f32, carry h in f32, emit bf16 copies for next-step GEMMs.
// 128 waves total: wave g -> m_tile (2) x j_tile (64 cols of H).
// Six f32 accumulators per wave (r/z/n for gi and gh).
// ---------------------------------------------------------------------------
__global__ __launch_bounds__(256) void gru_cell_kernel(
    const __bf16* __restrict__ A0, int lenA0, int ldA0,
    const __bf16* __restrict__ A1, int ldA1, int Kin,
    const __bf16* __restrict__ hb_in, const float* __restrict__ hf_in,
    const __bf16* __restrict__ Wih, const __bf16* __restrict__ Whh,
    const float* __restrict__ bih, const float* __restrict__ bhh,
    float* __restrict__ hf_out, __bf16* __restrict__ hb_out,
    __bf16* __restrict__ ctx_out, __bf16* __restrict__ allh1, int s) {
  int tid  = threadIdx.x;
  int wave = tid >> 5;
  int lane = tid & 31;
  int g    = blockIdx.x * 8 + wave;   // 0..127
  int m_t  = g & 1;
  int j_t  = g >> 1;                  // 0..63
  int mbase = m_t * 16;
  int jbase = j_t * 16;
  int ln = lane & 15;
  int lh = lane >> 4;

  v8f air = {}, aiz = {}, ain = {};
  v8f ahr = {}, ahz = {}, ahn = {};

  // ---- gi = x @ Wih^T ----
  for (int kb = 0; kb < Kin; kb += 32) {
    const __bf16* arow;
    if (kb < lenA0) arow = A0 + (size_t)(mbase + ln) * ldA0 + kb;
    else            arow = A1 + (size_t)(mbase + ln) * ldA1 + (kb - lenA0);
    v16bf a = load_frag16(arow + 16 * lh, arow + 16 * lh + 8);

    const __bf16* b0 = Wih + (size_t)(jbase + ln) * Kin + kb + 16 * lh;
    const __bf16* b1 = b0 + (size_t)HID * Kin;
    const __bf16* b2 = b0 + (size_t)(2 * HID) * Kin;
    air = wmma_bf16(air, a, load_frag16(b0, b0 + 8));
    aiz = wmma_bf16(aiz, a, load_frag16(b1, b1 + 8));
    ain = wmma_bf16(ain, a, load_frag16(b2, b2 + 8));
  }

  // ---- gh = h @ Whh^T ----
  for (int kb = 0; kb < HID; kb += 32) {
    const __bf16* arow = hb_in + (size_t)(mbase + ln) * HID + kb;
    v16bf a = load_frag16(arow + 16 * lh, arow + 16 * lh + 8);

    const __bf16* b0 = Whh + (size_t)(jbase + ln) * HID + kb + 16 * lh;
    const __bf16* b1 = b0 + (size_t)HID * HID;
    const __bf16* b2 = b0 + (size_t)(2 * HID) * HID;
    ahr = wmma_bf16(ahr, a, load_frag16(b0, b0 + 8));
    ahz = wmma_bf16(ahz, a, load_frag16(b1, b1 + 8));
    ahn = wmma_bf16(ahn, a, load_frag16(b2, b2 + 8));
  }

  // ---- gates + carry ----
  int col = jbase + ln;
  float bir = bih[col], biz = bih[HID + col], bin = bih[2 * HID + col];
  float bhr = bhh[col], bhz = bhh[HID + col], bhn = bhh[2 * HID + col];

#pragma unroll
  for (int r = 0; r < 8; ++r) {
    int m   = mbase + 8 * lh + r;          // batch row (C/D layout)
    float rg = sigmoidf_(air[r] + bir + ahr[r] + bhr);
    float zg = sigmoidf_(aiz[r] + biz + ahz[r] + bhz);
    float ng = tanhf_(ain[r] + bin + rg * (ahn[r] + bhn));
    float hp = hf_in[(size_t)m * HID + col];
    float h  = (1.0f - zg) * ng + zg * hp;
    hf_out[(size_t)m * HID + col] = h;
    __bf16 hb = (__bf16)h;
    hb_out[(size_t)m * HID + col] = hb;
    if (ctx_out) ctx_out[(size_t)m * HID + col] = hb;
    // all_h1 in [b][s][k] layout so the vocab GEMM writes s-contiguous output
    if (allh1)   allh1[((size_t)m * SEQLEN + s) * HID + col] = hb;
  }
}

// ---------------------------------------------------------------------------
// Hoisted vocab projection: logits[(b,s), v] = all_h1 @ Wv^T + bv
// M=2048 (rows = b*64+s), N=32000, K=1024.
// Each wave computes a 32x64 tile (2 M-tiles x 4 N-tiles, 8 f32 accumulators):
// per 32-K step, 6 fragment loads feed 8 WMMAs (B fragments amortized over
// two M-tiles), halving L2 fragment traffic vs a 16x64 tile.
// Writes directly into d_out in [B][V][S] layout: C element r maps to
// s = sbase + 8*lh + r, so each acc stores as two contiguous float4s.
// ---------------------------------------------------------------------------
__global__ __launch_bounds__(256) void vocab_gemm_kernel(
    const __bf16* __restrict__ A, const __bf16* __restrict__ W,
    const float* __restrict__ bv, float* __restrict__ out) {
  int tid  = threadIdx.x;
  int wave = tid >> 5;
  int lane = tid & 31;
  int g    = blockIdx.x * 8 + wave;   // 0..31999
  int n_t  = g % 500;
  int m_t  = g / 500;                 // 0..63
  int mbase = m_t * 32;
  long nbase = (long)n_t * 64;
  int ln = lane & 15;
  int lh = lane >> 4;

  v8f acc0[4] = {};
  v8f acc1[4] = {};
  for (int kb = 0; kb < HID; kb += 32) {
    const __bf16* arow0 = A + (size_t)(mbase + ln) * HID + kb + 16 * lh;
    const __bf16* arow1 = arow0 + (size_t)16 * HID;
    v16bf a0 = load_frag16(arow0, arow0 + 8);
    v16bf a1 = load_frag16(arow1, arow1 + 8);
#pragma unroll
    for (int c = 0; c < 4; ++c) {
      const __bf16* b0 = W + (size_t)(nbase + c * 16 + ln) * HID + kb + 16 * lh;
      v16bf b = load_frag16(b0, b0 + 8);
      acc0[c] = wmma_bf16(acc0[c], a0, b);
      acc1[c] = wmma_bf16(acc1[c], a1, b);
    }
  }

#pragma unroll
  for (int i = 0; i < 2; ++i) {
    int mb    = mbase + i * 16;
    int b     = mb >> 6;        // 64 rows (s) per batch entry
    int sbase = mb & 63;
#pragma unroll
    for (int c = 0; c < 4; ++c) {
      v8f acc = i ? acc1[c] : acc0[c];
      long v = nbase + c * 16 + ln;
      float bb = bv[v];
      float* o = out + (long)b * VOCAB * SEQLEN + v * SEQLEN + sbase + 8 * lh;
      float4 f0 = make_float4(acc[0] + bb, acc[1] + bb, acc[2] + bb, acc[3] + bb);
      float4 f1 = make_float4(acc[4] + bb, acc[5] + bb, acc[6] + bb, acc[7] + bb);
      *reinterpret_cast<float4*>(o)     = f0;
      *reinterpret_cast<float4*>(o + 4) = f1;
    }
  }
}

// ---------------------------------------------------------------------------
// In-place log_softmax over V for each (b, s). Block = one b; threadIdx.x&63
// indexes s so reads/writes of out[b][v][s..s+63] are coalesced.
// ---------------------------------------------------------------------------
__global__ __launch_bounds__(1024) void logsoftmax_kernel(float* __restrict__ out) {
  __shared__ float red[16][64];
  __shared__ float mx_s[64];
  __shared__ float lse_s[64];
  int b  = blockIdx.x;
  int tx = threadIdx.x & 63;   // s
  int ty = threadIdx.x >> 6;   // 0..15
  float* base = out + (size_t)b * VOCAB * SEQLEN;

  float mx = -3.4e38f;
  for (int v = ty; v < VOCAB; v += 16)
    mx = fmaxf(mx, base[(size_t)v * SEQLEN + tx]);
  red[ty][tx] = mx;
  __syncthreads();
  if (ty == 0) {
    float m = red[0][tx];
    for (int i = 1; i < 16; ++i) m = fmaxf(m, red[i][tx]);
    mx_s[tx] = m;
  }
  __syncthreads();
  float m = mx_s[tx];

  float sum = 0.0f;
  for (int v = ty; v < VOCAB; v += 16)
    sum += __expf(base[(size_t)v * SEQLEN + tx] - m);
  __syncthreads();
  red[ty][tx] = sum;
  __syncthreads();
  if (ty == 0) {
    float ssum = 0.0f;
    for (int i = 0; i < 16; ++i) ssum += red[i][tx];
    lse_s[tx] = m + __logf(ssum);
  }
  __syncthreads();
  float l = lse_s[tx];
  for (int v = ty; v < VOCAB; v += 16)
    base[(size_t)v * SEQLEN + tx] -= l;
}

// ---------------------------------------------------------------------------
extern "C" void kernel_launch(void* const* d_in, const int* in_sizes, int n_in,
                              void* d_out, int out_size, void* d_ws, size_t ws_size,
                              hipStream_t stream) {
  (void)in_sizes; (void)n_in; (void)out_size; (void)ws_size;

  const int*   idx  = (const int*)  d_in[0];
  // d_in[1] seq_lengths, d_in[2] seq_enc_states, d_in[3] enc_padding_mask: unused
  const float* ench = (const float*)d_in[4];
  const float* emb  = (const float*)d_in[5];
  const float* Wih0 = (const float*)d_in[6];
  const float* Whh0 = (const float*)d_in[7];
  const float* bih0 = (const float*)d_in[8];
  const float* bhh0 = (const float*)d_in[9];
  const float* Wih1 = (const float*)d_in[10];
  const float* Whh1 = (const float*)d_in[11];
  const float* bih1 = (const float*)d_in[12];
  const float* bhh1 = (const float*)d_in[13];
  const float* Wv   = (const float*)d_in[14];
  const float* bv   = (const float*)d_in[15];
  float* out = (float*)d_out;

  char* ws = (char*)d_ws;
  size_t off = 0;
  auto alloc = [&](size_t bytes) -> void* {
    void* p = ws + off;
    off = (off + bytes + 255) & ~(size_t)255;
    return p;
  };

  __bf16* Wv_b   = (__bf16*)alloc((size_t)VOCAB * HID * 2);
  __bf16* Wih0_b = (__bf16*)alloc((size_t)3 * HID * (HID + EMB) * 2);
  __bf16* Whh0_b = (__bf16*)alloc((size_t)3 * HID * HID * 2);
  __bf16* Wih1_b = (__bf16*)alloc((size_t)3 * HID * HID * 2);
  __bf16* Whh1_b = (__bf16*)alloc((size_t)3 * HID * HID * 2);
  __bf16* xemb   = (__bf16*)alloc((size_t)SEQLEN * BATCH * EMB * 2);
  __bf16* allh1  = (__bf16*)alloc((size_t)BATCH * SEQLEN * HID * 2);
  float*  h0f[2] = {(float*)alloc(BATCH * HID * 4), (float*)alloc(BATCH * HID * 4)};
  float*  h1f[2] = {(float*)alloc(BATCH * HID * 4), (float*)alloc(BATCH * HID * 4)};
  __bf16* h0b[2] = {(__bf16*)alloc(BATCH * HID * 2), (__bf16*)alloc(BATCH * HID * 2)};
  __bf16* h1b[2] = {(__bf16*)alloc(BATCH * HID * 2), (__bf16*)alloc(BATCH * HID * 2)};
  __bf16* ctx[2] = {(__bf16*)alloc(BATCH * HID * 2), (__bf16*)alloc(BATCH * HID * 2)};

  auto cvt = [&](const float* src, __bf16* dst, long n) {
    long blk = (n + 255) / 256;
    f32_to_bf16_kernel<<<dim3((unsigned)blk), dim3(256), 0, stream>>>(src, dst, n);
  };
  cvt(Wv,   Wv_b,   (long)VOCAB * HID);
  cvt(Wih0, Wih0_b, (long)3 * HID * (HID + EMB));
  cvt(Whh0, Whh0_b, (long)3 * HID * HID);
  cvt(Wih1, Wih1_b, (long)3 * HID * HID);
  cvt(Whh1, Whh1_b, (long)3 * HID * HID);

  {
    long n = (long)SEQLEN * BATCH * EMB;
    embed_kernel<<<dim3((unsigned)((n + 255) / 256)), dim3(256), 0, stream>>>(idx, emb, xemb);
  }
  init_state_kernel<<<dim3((BATCH * HID + 255) / 256), dim3(256), 0, stream>>>(
      ench, h0f[0], h0b[0], h1f[0], h1b[0], ctx[0]);

  // ---- sequential GRU recurrence (weights hot in L2 across steps) ----
  for (int s = 0; s < SEQLEN; ++s) {
    int p = s & 1, np = p ^ 1;
    // layer 0: x = [xemb[s] | ctx]
    gru_cell_kernel<<<dim3(16), dim3(256), 0, stream>>>(
        xemb + (size_t)s * BATCH * EMB, EMB, EMB, ctx[p], HID, HID + EMB,
        h0b[p], h0f[p], Wih0_b, Whh0_b, bih0, bhh0,
        h0f[np], h0b[np], (__bf16*)nullptr, (__bf16*)nullptr, s);
    // layer 1: x = h0_new; also emits ctx for next step and all_h1[s]
    gru_cell_kernel<<<dim3(16), dim3(256), 0, stream>>>(
        h0b[np], HID, HID, (const __bf16*)nullptr, HID, HID,
        h1b[p], h1f[p], Wih1_b, Whh1_b, bih1, bhh1,
        h1f[np], h1b[np], ctx[np], allh1, s);
  }

  // ---- hoisted vocab projection: one big WMMA GEMM (32x64 tiles/wave) ----
  vocab_gemm_kernel<<<dim3(4000), dim3(256), 0, stream>>>(allh1, Wv_b, bv, out);

  // ---- in-place log_softmax over V ----
  logsoftmax_kernel<<<dim3(BATCH), dim3(1024), 0, stream>>>(out);
}
